// GraphConvolution_49177375539507
// MI455X (gfx1250) — matrix operand order repaired and checked by reading, not compiled, not run on hardware
//
#include <hip/hip_runtime.h>

typedef _Float16 h4   __attribute__((ext_vector_type(4)));
typedef _Float16 h8   __attribute__((ext_vector_type(8)));
typedef _Float16 v16h __attribute__((ext_vector_type(16)));
typedef float    v8f  __attribute__((ext_vector_type(8)));
typedef float    f4   __attribute__((ext_vector_type(4)));

#define NB 512
#define NE 4
#define NN 256
#define NF 64

#define WMMA(a, b, c) \
  __builtin_amdgcn_wmma_f32_16x16x32_f16(false, (a), false, (b), (short)0, (c), false, false)

// A operand (16x32, f16) from row-major [row][k] LDS buffer, stride in f16 elems.
// ISA layout: lanes 0-15: M=lane, elems 0-7 -> K 0-7, elems 8-15 -> K 16-23
//             lanes 16-31: M=lane-16, elems 0-7 -> K 8-15, elems 8-15 -> K 24-31
__device__ __forceinline__ v16h lds_load_A(const _Float16* base, int stride,
                                           int row0, int k0, int lane) {
  const int m    = lane & 15;
  const int koff = (lane < 16) ? 0 : 8;
  const _Float16* p = base + (row0 + m) * stride + k0 + koff;
  h8 lo = *(const h8*)(p);
  h8 hi = *(const h8*)(p + 16);
  return __builtin_shufflevector(lo, hi, 0,1,2,3,4,5,6,7,8,9,10,11,12,13,14,15);
}

// B operand (32x16, f16) from TRANSPOSED [g][k] LDS buffer, stride in f16 elems.
// ISA layout: lane N=lane&15; elems 0-7 -> K = k0 + koff + i, elems 8-15 -> K = k0 + 16 + koff + (i-8), koff = (lane<16 ? 0 : 8)
// Two 16B-aligned h8 loads (2x ds_load_b128).
__device__ __forceinline__ v16h lds_load_B(const _Float16* baseT, int stride,
                                           int g0, int k0, int lane) {
  const int g = lane & 15;
  const int k = k0 + ((lane < 16) ? 0 : 8);
  const _Float16* p = baseT + (g0 + g) * stride + k;
  h8 lo = *(const h8*)(p);
  h8 hi = *(const h8*)(p + 16);
  return __builtin_shufflevector(lo, hi, 0,1,2,3,4,5,6,7,8,9,10,11,12,13,14,15);
}

// Raw f32 fetch of an A tile (16x32) from global row-major [row][k], stride NN.
__device__ __forceinline__ void glo_fetch_A(const float* base, int row0, int k0,
                                            int lane, f4 (&r)[4]) {
  const int m    = lane & 15;
  const int koff = (lane < 16) ? 0 : 8;
  const float* p = base + (size_t)(row0 + m) * NN + k0 + koff;
  r[0] = *(const f4*)(p);
  r[1] = *(const f4*)(p + 4);
  r[2] = *(const f4*)(p + 16);
  r[3] = *(const f4*)(p + 20);
}

__device__ __forceinline__ v16h cvt_A(const f4 (&r)[4]) {
  v16h o;
#pragma unroll
  for (int i = 0; i < 4; ++i) {
    o[i]      = (_Float16)r[0][i];
    o[4 + i]  = (_Float16)r[1][i];
    o[8 + i]  = (_Float16)r[2][i];
    o[12 + i] = (_Float16)r[3][i];
  }
  return o;
}

__global__ __launch_bounds__(256, 1)
void gcn_fused_kernel(const float* __restrict__ node, const float* __restrict__ adj,
                      const float* __restrict__ W_edge, const float* __restrict__ b_edge,
                      const float* __restrict__ W_add, const float* __restrict__ b_add,
                      float* __restrict__ out) {
  __shared__ _Float16 s_node[NN * NF];            // 32 KB  [n][f]
  __shared__ _Float16 s_w[(NE + 1) * NF * NF];    // 40 KB  [mat][g][f] (transposed)
  __shared__ _Float16 s_hid[NE * NF * NN];        // 128 KB [e][g][m]  (transposed)

  const int b    = blockIdx.x;
  const int t    = threadIdx.x;
  const int lane = t & 31;
  const int wv   = t >> 5;       // 8 waves; wave wv owns output rows [wv*32, wv*32+32)

  // ---------------- Phase 0: stage node (f16) and transposed weights (f16) ----------------
  const float* nodeB = node + (size_t)b * NN * NF;
#pragma unroll
  for (int idx = t; idx < NN * NF / 4; idx += 256) {
    f4 v = *((const f4*)nodeB + idx);
    h4 h;
#pragma unroll
    for (int i = 0; i < 4; ++i) h[i] = (_Float16)v[i];
    *(h4*)(&s_node[idx * 4]) = h;
  }
#pragma unroll
  for (int m5 = 0; m5 < NE + 1; ++m5) {
    const float* src = (m5 < NE) ? (W_edge + m5 * NF * NF) : W_add;
    for (int idx = t; idx < NF * NF; idx += 256) {
      const int f = idx >> 6, g = idx & 63;
      s_w[m5 * NF * NF + g * NF + f] = (_Float16)src[idx];   // transpose: [g][f]
    }
  }
  __syncthreads();

  // Node A operands are invariant across all 5 weight matrices: load once, keep in VGPRs.
  const int row0w = wv * 32;
  v16h na[2][2];
#pragma unroll
  for (int rt2 = 0; rt2 < 2; ++rt2) {
    na[rt2][0] = lds_load_A(s_node, NF, row0w + rt2 * 16, 0, lane);
    na[rt2][1] = lds_load_A(s_node, NF, row0w + rt2 * 16, 32, lane);
  }

  // ---------------- Phase 1: hidden[e] = node@W_e + b_e  (store transposed f16) ----------------
#pragma unroll 1
  for (int e = 0; e < NE; ++e) {
#pragma unroll
    for (int ct = 0; ct < 4; ++ct) {
      // each W B-tile loaded exactly once, feeds both row-tiles
      v16h b0 = lds_load_B(s_w + e * NF * NF, NF, ct * 16, 0, lane);
      v16h b1 = lds_load_B(s_w + e * NF * NF, NF, ct * 16, 32, lane);
      const float bias = b_edge[e * NF + ct * 16 + (lane & 15)];
#pragma unroll
      for (int rt2 = 0; rt2 < 2; ++rt2) {
        const int row0 = row0w + rt2 * 16;
        v8f h = {};
        h = WMMA(na[rt2][0], b0, h);
        h = WMMA(na[rt2][1], b1, h);
        _Float16* p = s_hid + (e * NF + ct * 16 + (lane & 15)) * NN + row0 + ((lane < 16) ? 0 : 8);
        h8 hh;
#pragma unroll
        for (int r = 0; r < 8; ++r) hh[r] = (_Float16)(h[r] + bias);
        *(h8*)p = hh;   // contiguous 16B: stores [g][m..m+7]
      }
    }
  }

  // ---------------- Residual node@W_add directly into final accumulators ----------------
  v8f acc[2][4];
#pragma unroll
  for (int ct = 0; ct < 4; ++ct) {
    v16h b0 = lds_load_B(s_w + NE * NF * NF, NF, ct * 16, 0, lane);
    v16h b1 = lds_load_B(s_w + NE * NF * NF, NF, ct * 16, 32, lane);
#pragma unroll
    for (int rt2 = 0; rt2 < 2; ++rt2) {
      v8f c = {};
      c = WMMA(na[rt2][0], b0, c);
      c = WMMA(na[rt2][1], b1, c);
      acc[rt2][ct] = c;
    }
  }
  __syncthreads();

  // ---------------- Phase 2: sum_e adj[b,e] @ hidden[e] ----------------
  // adj streamed from HBM exactly once, one-deep software pipeline:
  // raw f32 tiles for iteration it+1 are in flight while iteration it's 8 WMMAs run.
  const float* adjB = adj + (size_t)b * NE * NN * NN;
  f4 raw0[4], raw1[4];
  glo_fetch_A(adjB, row0w, 0, lane, raw0);
  glo_fetch_A(adjB, row0w + 16, 0, lane, raw1);

#pragma unroll 1
  for (int it = 0; it < NE * 8; ++it) {
    const int e  = it >> 3;
    const int k0 = (it & 7) * 32;

    v16h aA0 = cvt_A(raw0);
    v16h aA1 = cvt_A(raw1);

    // prefetch next (clamped: last iteration redundantly refetches itself)
    const int itn = (it < NE * 8 - 1) ? (it + 1) : it;
    const float* adjN = adjB + (size_t)(itn >> 3) * NN * NN;
    const int k0n = (itn & 7) * 32;
    glo_fetch_A(adjN, row0w, k0n, lane, raw0);
    glo_fetch_A(adjN, row0w + 16, k0n, lane, raw1);

    const _Float16* hidE = s_hid + e * NF * NN;
    v16h bb0 = lds_load_B(hidE, NN, 0,  k0, lane);
    v16h bb1 = lds_load_B(hidE, NN, 16, k0, lane);
    v16h bb2 = lds_load_B(hidE, NN, 32, k0, lane);
    v16h bb3 = lds_load_B(hidE, NN, 48, k0, lane);

    acc[0][0] = WMMA(aA0, bb0, acc[0][0]);
    acc[1][0] = WMMA(aA1, bb0, acc[1][0]);
    acc[0][1] = WMMA(aA0, bb1, acc[0][1]);
    acc[1][1] = WMMA(aA1, bb1, acc[1][1]);
    acc[0][2] = WMMA(aA0, bb2, acc[0][2]);
    acc[1][2] = WMMA(aA1, bb2, acc[1][2]);
    acc[0][3] = WMMA(aA0, bb3, acc[0][3]);
    acc[1][3] = WMMA(aA1, bb3, acc[1][3]);
  }

  // ---------------- Writeout + b_add ----------------
  float* outB = out + (size_t)b * NN * NF;
#pragma unroll
  for (int rt2 = 0; rt2 < 2; ++rt2) {
    const int rowb = row0w + rt2 * 16 + ((lane < 16) ? 0 : 8);
#pragma unroll
    for (int ct = 0; ct < 4; ++ct) {
      const int g = ct * 16 + (lane & 15);
      const float bias = b_add[g];
#pragma unroll
      for (int r = 0; r < 8; ++r)
        outB[(rowb + r) * NF + g] = acc[rt2][ct][r] + bias;
    }
  }
}

extern "C" void kernel_launch(void* const* d_in, const int* in_sizes, int n_in,
                              void* d_out, int out_size, void* d_ws, size_t ws_size,
                              hipStream_t stream) {
  (void)in_sizes; (void)n_in; (void)d_ws; (void)ws_size; (void)out_size;
  const float* node   = (const float*)d_in[0];
  const float* adj    = (const float*)d_in[1];
  const float* W_edge = (const float*)d_in[2];
  const float* b_edge = (const float*)d_in[3];
  const float* W_add  = (const float*)d_in[4];
  const float* b_add  = (const float*)d_in[5];
  float* out = (float*)d_out;

  hipLaunchKernelGGL(gcn_fused_kernel, dim3(NB), dim3(256), 0, stream,
                     node, adj, W_edge, b_edge, W_add, b_add, out);
}
